// RNNODEEncoder_19902878450161
// MI455X (gfx1250) — matrix operand order repaired and checked by reading, not compiled, not run on hardware
//
#include <hip/hip_runtime.h>
#include <hip/hip_bf16.h>
#include <math.h>

#define T_LEN 512
#define B_SZ  256
#define H_SZ  256
#define NLAY  4

// LDS layout (dynamic shared, so byte offsets == DS addresses)
#define OFF_Y0     0        // 16 KB  f32 [16][256]
#define OFF_KB     16384    // 96 KB  f32 6 x [16][256]  (reused as gi/gh [16][768]x2)
#define OFF_HBUF   114688   // 16 KB  f32 [16][256]
#define OFF_ABUF   131072   // 8448 B bf16 [16][264]
#define OFF_XSTG   139520   // 16 KB  f32 [16][256]  (async x staging)
#define OFF_LNG    155904   // 4 KB   f32 [4][256]
#define OFF_LNB    160000   // 4 KB   f32 [4][256]
#define OFF_DT     164096
#define OFF_TM     164160
#define OFF_MS     164224
#define SMEM_BYTES 164352

typedef __attribute__((ext_vector_type(16))) __bf16 v16bf;
typedef __attribute__((ext_vector_type(8)))  float  v8f;

// Tsit5 stage coefficients: row r = coefficients for stage r+2 (padded with 0)
__constant__ float TSA[5][5] = {
  {0.161f, 0.f, 0.f, 0.f, 0.f},
  {-0.008480655492356989f, 0.335480655492357f, 0.f, 0.f, 0.f},
  {2.8971530571054935f, -6.359448489975075f, 4.3622954328695815f, 0.f, 0.f},
  {5.325864828439257f, -11.748883564062828f, 7.4955393428898365f, -0.09249506636175525f, 0.f},
  {5.86145544294642f, -12.92096931784711f, 8.159367898576159f, -0.071584973281401f, -0.028269050394068383f},
};
__constant__ float TSB[6] = {0.09646076681806523f, 0.01f, 0.4798896504144996f,
                             1.379008574103742f, -3.290069515436081f, 2.324710524099774f};

__device__ __forceinline__ unsigned short f32_to_bf16(float f) {
  union { float f; unsigned int u; } v; v.f = f;
  unsigned int r = v.u + 0x7FFFu + ((v.u >> 16) & 1u);   // round-to-nearest-even
  return (unsigned short)(r >> 16);
}
__device__ __forceinline__ float softplus_f(float x) {
  return (x > 20.f) ? x : log1pf(__expf(x));
}
__device__ __forceinline__ float sigmoid_f(float x) {
  return 1.f / (1.f + __expf(-x));
}

// ---------------------------------------------------------------------------
// Weight packing: fp32 [K,N] (or transposed [N,K]) -> bf16 WMMA-B fragments.
// Fragment f = nTile*(K/32) + kTile holds a 32x16 (KxN) slab; lane l keeps
// n = nTileBase + (l&15), k = kTileBase + (l>=16?16:0) + j, j = 0..15
// (contiguous k order, same as the A operand) -> two contiguous 16B loads.
// ---------------------------------------------------------------------------
__global__ void pack_kernel(const float* __restrict__ src, unsigned short* __restrict__ dst,
                            int K, int N, int transpose) {
  int tid = blockIdx.x * 256 + threadIdx.x;
  int total = K * N;
  if (tid >= total) return;
  int frag = tid >> 9;
  int rem  = tid & 511;
  int lane = rem >> 4;
  int j    = rem & 15;
  int kTiles = K >> 5;
  int nTile = frag / kTiles;
  int kTile = frag - nTile * kTiles;
  int n = nTile * 16 + (lane & 15);
  int k = kTile * 32 + ((lane >= 16) ? 16 : 0) + j;
  float v = transpose ? src[n * K + k] : src[k * N + n];
  dst[tid] = f32_to_bf16(v);
}

// ---------------------------------------------------------------------------
// One wave computes a 16x16 output tile of  C = A(16x256 bf16, LDS) * B + init
// A layout per lane: row = lane&15, k = kt*32 + (lane>=16?16:0)
// + j  -> two contiguous 16B chunks at kbase and kbase+8.
// Loads stay inside the K-loop: the compiler hoists B into registers when the
// pressure budget allows (proven in round-1 codegen), and never spills.
// ---------------------------------------------------------------------------
union FragU { v16bf v; uint4 q[2]; };

__device__ __forceinline__ void wave_gemm_tile(
    const unsigned short* __restrict__ wpk,   // packed weights (global)
    const unsigned short* abuf,               // LDS bf16 A, stride 264 halves
    const float* resid,                       // LDS f32 [16][256] or nullptr
    const float* __restrict__ bias,           // global bias[n] or nullptr
    float* outbuf, int ostride,               // LDS f32 output
    int nt, int lane) {
  const int n0   = nt * 16 + (lane & 15);
  const int hi   = lane >> 4;
  const int row  = lane & 15;
  const int koff = hi * 16;
  v8f c;
  const float bv = bias ? bias[n0] : 0.f;
#pragma unroll
  for (int i = 0; i < 8; ++i) {
    const int m = i + hi * 8;
    float v = bv;
    if (resid) v += resid[m * 256 + n0];
    c[i] = v;
  }
  const uint4* bq = (const uint4*)wpk + (size_t)nt * 512 + lane * 2;
#pragma unroll
  for (int kt = 0; kt < 8; ++kt) {
    FragU a, b;
    const int hb = row * 264 + kt * 32 + koff;
    a.q[0] = *(const uint4*)(abuf + hb);
    a.q[1] = *(const uint4*)(abuf + hb + 8);
    b.q[0] = bq[kt * 64];
    b.q[1] = bq[kt * 64 + 1];
    c = __builtin_amdgcn_wmma_f32_16x16x32_bf16(false, a.v, false, b.v,
                                                (short)0, c, false, false);
  }
#pragma unroll
  for (int i = 0; i < 8; ++i) {
    const int m = i + hi * 8;
    outbuf[m * ostride + n0] = c[i];
  }
}

// ---------------------------------------------------------------------------
// Persistent kernel: block b owns batch rows [16b,16b+16) for all 511 steps.
// ---------------------------------------------------------------------------
__global__ __launch_bounds__(512, 1)
void rnnode_kernel(const float* __restrict__ x, const float* __restrict__ timep,
                   const int* __restrict__ maskp,
                   const float* __restrict__ b_ih, const float* __restrict__ b_hh,
                   const float* __restrict__ b_in, const float* __restrict__ b_blk,
                   const float* __restrict__ ln_g, const float* __restrict__ ln_b,
                   const float* __restrict__ b_out,
                   const unsigned short* __restrict__ pk_wih,
                   const unsigned short* __restrict__ pk_whh,
                   const unsigned short* __restrict__ pk_win,
                   const unsigned short* __restrict__ pk_wblk,
                   const unsigned short* __restrict__ pk_wout,
                   float* __restrict__ out) {
  extern __shared__ char smem[];
  float* y0   = (float*)(smem + OFF_Y0);
  float* kb   = (float*)(smem + OFF_KB);
  float* hbuf = (float*)(smem + OFF_HBUF);
  unsigned short* abuf = (unsigned short*)(smem + OFF_ABUF);
  float* xstg = (float*)(smem + OFF_XSTG);
  float* lngs = (float*)(smem + OFF_LNG);
  float* lnbs = (float*)(smem + OFF_LNB);
  float* dtsh = (float*)(smem + OFF_DT);
  float* tmsh = (float*)(smem + OFF_TM);
  int*   msh  = (int*)  (smem + OFF_MS);

  const int tid   = threadIdx.x;
  const int lane  = tid & 31;
  const int wave  = tid >> 5;
  const int bBase = blockIdx.x * 16;

  // Init: h0 = 0, cache LN params in LDS
  for (int e = tid; e < 4096; e += 512) y0[e] = 0.f;
  for (int e = tid; e < 1024; e += 512) { lngs[e] = ln_g[e]; lnbs[e] = ln_b[e]; }
  __syncthreads();

  for (int s = 0; s < T_LEN - 1; ++s) {
    // -- async-stage this step's x tile into LDS; Tsit5 hides the latency --
    {
      const float* xr = x + ((size_t)(T_LEN - 2 - s) * B_SZ + bBase) * H_SZ;
      unsigned la0 = (unsigned)(OFF_XSTG) + (unsigned)tid * 16u;
      unsigned long long ga0 = (unsigned long long)(const char*)xr + (unsigned long long)tid * 16u;
      asm volatile("global_load_async_to_lds_b128 %0, %1, off"
                   :: "v"(la0), "v"(ga0) : "memory");
      unsigned la1 = la0 + 8192u;
      unsigned long long ga1 = ga0 + 8192ull;
      asm volatile("global_load_async_to_lds_b128 %0, %1, off"
                   :: "v"(la1), "v"(ga1) : "memory");
    }

    if (tid < 16) {
      const int b  = bBase + tid;
      const float t0 = timep[(size_t)(T_LEN - 1 - s) * B_SZ + b];
      const float t1 = timep[(size_t)(T_LEN - 2 - s) * B_SZ + b];
      const float tm = (t1 == t0) ? 1.f : 0.f;
      dtsh[tid] = (t1 - 1e-4f * tm) - t0;
      tmsh[tid] = tm;
      msh[tid]  = maskp[(size_t)(T_LEN - 2 - s) * B_SZ + b];
    }
    __syncthreads();

    // ================= Tsit5: six ODE evaluations =================
    for (int st = 0; st < 6; ++st) {
      // stage input -> abuf:  bf16( y0 + dt * sum_i a[st][i]*k_i )
      for (int e = tid; e < 4096; e += 512) {
        const int r = e >> 8, c = e & 255;
        float acc = 0.f;
        for (int i = 0; i < st; ++i) acc += TSA[st - 1][i] * kb[i * 4096 + e];
        abuf[r * 264 + c] = f32_to_bf16(y0[e] + dtsh[r] * acc);
      }
      __syncthreads();

      // ----- _ode_f(abuf) -> kb[st] -----
      wave_gemm_tile(pk_win, abuf, nullptr, b_in, hbuf, 256, wave, lane);
      __syncthreads();
      for (int blk = 0; blk < NLAY; ++blk) {
        // pull this block's packed-weight slice into near caches
        __builtin_prefetch(pk_wblk + blk * 65536 + wave * 4096 + lane * 128, 0, 3);
        { // pre-norm LayerNorm + softplus on row `wave`
          const int r = wave;
          float s1 = 0.f, s2 = 0.f;
          for (int c = lane; c < 256; c += 32) {
            const float v = hbuf[r * 256 + c]; s1 += v; s2 += v * v;
          }
          for (int off = 16; off; off >>= 1) {
            s1 += __shfl_xor(s1, off, 32);
            s2 += __shfl_xor(s2, off, 32);
          }
          const float mu  = s1 * (1.f / 256.f);
          const float var = fmaxf(s2 * (1.f / 256.f) - mu * mu, 0.f);
          const float rs  = rsqrtf(var + 1e-5f);
          for (int c = lane; c < 256; c += 32) {
            const float u = (hbuf[r * 256 + c] - mu) * rs * lngs[blk * 256 + c]
                            + lnbs[blk * 256 + c];
            abuf[r * 264 + c] = f32_to_bf16(softplus_f(u));
          }
        }
        __syncthreads();
        // h = h + u @ W_blk[blk] + b_blk[blk]   (residual via C-init)
        wave_gemm_tile(pk_wblk + blk * 65536, abuf, hbuf, b_blk + blk * 256,
                       hbuf, 256, wave, lane);
        __syncthreads();
      }
      for (int e = tid; e < 4096; e += 512)
        abuf[(e >> 8) * 264 + (e & 255)] = f32_to_bf16(hbuf[e]);
      __syncthreads();
      wave_gemm_tile(pk_wout, abuf, nullptr, b_out, kb + st * 4096, 256, wave, lane);
      __syncthreads();
    }

    // ======= combine + tmask select:  y0 := h_p =======
    for (int e = tid; e < 4096; e += 512) {
      const int r = e >> 8;
      float acc = 0.f;
      for (int i = 0; i < 6; ++i) acc += TSB[i] * kb[i * 4096 + e];
      const float yn = y0[e] + dtsh[r] * acc;
      y0[e] = (tmsh[r] != 0.f) ? y0[e] : yn;
    }
    __syncthreads();

    // ================= GRU cell =================
    asm volatile("s_wait_asynccnt 0" ::: "memory");  // x tile resident in LDS
    __syncthreads();
    for (int e = tid; e < 4096; e += 512)
      abuf[(e >> 8) * 264 + (e & 255)] = f32_to_bf16(xstg[e]);
    __syncthreads();
    float* gi = kb;            // [16][768] f32 (reuses dead k buffers)
    float* gh = kb + 12288;    // [16][768] f32
    for (int q = 0; q < 3; ++q)   // gi = x_t @ W_ih^T + b_ih
      wave_gemm_tile(pk_wih, abuf, nullptr, b_ih, gi, 768, wave + 16 * q, lane);
    __syncthreads();
    for (int e = tid; e < 4096; e += 512)
      abuf[(e >> 8) * 264 + (e & 255)] = f32_to_bf16(y0[e]);
    __syncthreads();
    for (int q = 0; q < 3; ++q)   // gh = h_p @ W_hh^T + b_hh
      wave_gemm_tile(pk_whh, abuf, nullptr, b_hh, gh, 768, wave + 16 * q, lane);
    __syncthreads();
    {
      float* orow = out + (size_t)s * (B_SZ * H_SZ) + (size_t)bBase * H_SZ;
      for (int e = tid; e < 4096; e += 512) {
        const int r = e >> 8, c = e & 255;
        const float hp = y0[e];
        const float rg = sigmoid_f(gi[r * 768 + c] + gh[r * 768 + c]);
        const float zg = sigmoid_f(gi[r * 768 + 256 + c] + gh[r * 768 + 256 + c]);
        const float ng = tanhf(gi[r * 768 + 512 + c] + rg * gh[r * 768 + 512 + c]);
        const float ht = msh[r] ? ((1.f - zg) * ng + zg * hp) : hp;
        y0[e]   = ht;
        orow[e] = ht;
      }
    }
    __syncthreads();
  }

  // h_last
  {
    float* hl = out + (size_t)(T_LEN - 1) * (B_SZ * H_SZ) + (size_t)bBase * H_SZ;
    for (int e = tid; e < 4096; e += 512) hl[e] = y0[e];
  }
}

extern "C" void kernel_launch(void* const* d_in, const int* in_sizes, int n_in,
                              void* d_out, int out_size, void* d_ws, size_t ws_size,
                              hipStream_t stream) {
  (void)in_sizes; (void)n_in; (void)out_size; (void)ws_size;
  const float* x     = (const float*)d_in[0];
  const float* timep = (const float*)d_in[1];
  const int*   maskp = (const int*)  d_in[2];
  const float* W_ih  = (const float*)d_in[3];
  const float* W_hh  = (const float*)d_in[4];
  const float* b_ih  = (const float*)d_in[5];
  const float* b_hh  = (const float*)d_in[6];
  const float* W_in  = (const float*)d_in[7];
  const float* b_in  = (const float*)d_in[8];
  const float* W_blk = (const float*)d_in[9];
  const float* b_blk = (const float*)d_in[10];
  const float* ln_g  = (const float*)d_in[11];
  const float* ln_b  = (const float*)d_in[12];
  const float* W_out = (const float*)d_in[13];
  const float* b_out = (const float*)d_in[14];

  unsigned short* pk      = (unsigned short*)d_ws;
  unsigned short* pk_wih  = pk;                // 256x768 (transposed W_ih)
  unsigned short* pk_whh  = pk + 196608;       // 256x768 (transposed W_hh)
  unsigned short* pk_win  = pk + 393216;       // 256x256
  unsigned short* pk_wblk = pk + 458752;       // 4 x 256x256
  unsigned short* pk_wout = pk + 720896;       // 256x256

  pack_kernel<<<dim3((196608 + 255) / 256), dim3(256), 0, stream>>>(W_ih, pk_wih, 256, 768, 1);
  pack_kernel<<<dim3((196608 + 255) / 256), dim3(256), 0, stream>>>(W_hh, pk_whh, 256, 768, 1);
  pack_kernel<<<dim3(256), dim3(256), 0, stream>>>(W_in, pk_win, 256, 256, 0);
  for (int i = 0; i < 4; ++i)
    pack_kernel<<<dim3(256), dim3(256), 0, stream>>>(W_blk + i * 65536,
                                                     pk_wblk + i * 65536, 256, 256, 0);
  pack_kernel<<<dim3(256), dim3(256), 0, stream>>>(W_out, pk_wout, 256, 256, 0);

  (void)hipFuncSetAttribute((const void*)rnnode_kernel,
                            hipFuncAttributeMaxDynamicSharedMemorySize,
                            (int)SMEM_BYTES);
  rnnode_kernel<<<dim3(16), dim3(512), SMEM_BYTES, stream>>>(
      x, timep, maskp, b_ih, b_hh, b_in, b_blk, ln_g, ln_b, b_out,
      pk_wih, pk_whh, pk_win, pk_wblk, pk_wout, (float*)d_out);
}
